// ScaledDotProductAttention_73375221285052
// MI455X (gfx1250) — matrix operand order repaired and checked by reading, not compiled, not run on hardware
//
#include <hip/hip_runtime.h>
#include <hip/hip_bf16.h>

// ---------------------------------------------------------------------------
// ScaledDotProductAttention for MI455X (gfx1250, wave32, WMMA)
//   S = (Q * 1/64)^T @ K          (bf16 WMMA, f32 acc)  -> ws (bf16)
//   P = softmax_rows(S)           (in-place in ws)
//   O = P @ V^T                   (bf16 WMMA, f32 acc)  -> d_out (f32)
// Double-buffered LDS + register prefetch, one barrier per K-step.
// ---------------------------------------------------------------------------

typedef __bf16 v16bf __attribute__((ext_vector_type(16)));
typedef float  v8f   __attribute__((ext_vector_type(8)));

#define N_ROWS 4096   // queries rows  (inner dim of GEMM1)
#define D_DIM  4096   // queries cols  (output rows)
#define M_DIM  8192   // keys/values cols (softmax axis, inner dim of GEMM2)
#define V_DIM  4096   // values rows   (output cols)

__device__ __forceinline__ unsigned short f32_to_bf16(float f) {
  union { __bf16 h; unsigned short u; } c;
  c.h = (__bf16)f;                     // RNE; hardware cvt if available
  return c.u;
}
__device__ __forceinline__ float bf16_to_f32(unsigned short h) {
  return __uint_as_float(((unsigned int)h) << 16);
}

union Frag { v16bf v; uint4 q[2]; };

// ---------------------------------------------------------------------------
// GEMM1: S[d][m] = sum_n bf16(Q[n][d]/64) * bf16(K[n][m]),  S stored bf16
// ---------------------------------------------------------------------------
__global__ __launch_bounds__(256) void gemm_qk(const float* __restrict__ Q,
                                               const float* __restrict__ Kmat,
                                               unsigned short* __restrict__ S) {
  __shared__ alignas(16) unsigned short As[2][128][40];   // [buf][d][kk]
  __shared__ alignas(16) unsigned short Bs[2][128][40];   // [buf][m][kk]

  const int tid    = threadIdx.x;
  const int lane   = tid & 31;
  const int laneLo = lane & 15;
  const int laneHi = lane >> 4;
  const int waveId = tid >> 5;
  const int wr     = waveId & 1;    // 2 wave-rows
  const int wc     = waveId >> 1;   // 4 wave-cols
  const int d0 = blockIdx.y * 128;
  const int m0 = blockIdx.x * 128;

  float4 rq[4], rk[4];
  // prologue: fetch K-slice 0 into registers
#pragma unroll
  for (int j = 0; j < 4; ++j) {
    const int idx = tid + 256 * j;
    const int kk  = idx >> 5;
    const int e4  = idx & 31;
    rq[j] = *(const float4*)(Q    + (size_t)kk * D_DIM + d0 + e4 * 4);
    rk[j] = *(const float4*)(Kmat + (size_t)kk * M_DIM + m0 + e4 * 4);
  }

  v8f acc[4][2] = {};
  int buf = 0;

  for (int k0 = 0; k0 < N_ROWS; k0 += 32) {
    // commit prefetched registers to LDS (bf16 convert; 1/64 scale on A, exact)
#pragma unroll
    for (int j = 0; j < 4; ++j) {
      const int idx = tid + 256 * j;
      const int kk  = idx >> 5;
      const int e4  = idx & 31;
      As[buf][e4 * 4 + 0][kk] = f32_to_bf16(rq[j].x * 0.015625f);
      As[buf][e4 * 4 + 1][kk] = f32_to_bf16(rq[j].y * 0.015625f);
      As[buf][e4 * 4 + 2][kk] = f32_to_bf16(rq[j].z * 0.015625f);
      As[buf][e4 * 4 + 3][kk] = f32_to_bf16(rq[j].w * 0.015625f);
      Bs[buf][e4 * 4 + 0][kk] = f32_to_bf16(rk[j].x);
      Bs[buf][e4 * 4 + 1][kk] = f32_to_bf16(rk[j].y);
      Bs[buf][e4 * 4 + 2][kk] = f32_to_bf16(rk[j].z);
      Bs[buf][e4 * 4 + 3][kk] = f32_to_bf16(rk[j].w);
    }
    __syncthreads();

    // issue global loads for the next K-slice (overlaps with WMMA below)
    if (k0 + 32 < N_ROWS) {
#pragma unroll
      for (int j = 0; j < 4; ++j) {
        const int idx = tid + 256 * j;
        const int kk  = idx >> 5;
        const int e4  = idx & 31;
        rq[j] = *(const float4*)(Q    + (size_t)(k0 + 32 + kk) * D_DIM + d0 + e4 * 4);
        rk[j] = *(const float4*)(Kmat + (size_t)(k0 + 32 + kk) * M_DIM + m0 + e4 * 4);
      }
    }

    Frag a[4], b[2];
#pragma unroll
    for (int mi = 0; mi < 4; ++mi) {
      const unsigned short* pa = &As[buf][wr * 64 + mi * 16 + laneLo][laneHi * 8];
      a[mi].q[0] = *(const uint4*)pa;          // K = klo .. klo+7
      a[mi].q[1] = *(const uint4*)(pa + 16);   // K = 16+klo .. 16+klo+7
    }
#pragma unroll
    for (int ni = 0; ni < 2; ++ni) {
      const unsigned short* pb = &Bs[buf][wc * 32 + ni * 16 + laneLo][laneHi * 8];
      b[ni].q[0] = *(const uint4*)pb;
      b[ni].q[1] = *(const uint4*)(pb + 16);
    }
#pragma unroll
    for (int mi = 0; mi < 4; ++mi)
#pragma unroll
      for (int ni = 0; ni < 2; ++ni)
        acc[mi][ni] = __builtin_amdgcn_wmma_f32_16x16x32_bf16(
            false, a[mi].v, false, b[ni].v, (short)0, acc[mi][ni], false, false);

    buf ^= 1;
  }

#pragma unroll
  for (int mi = 0; mi < 4; ++mi)
#pragma unroll
    for (int ni = 0; ni < 2; ++ni) {
      const int rowB = d0 + wr * 64 + mi * 16 + 8 * laneHi;
      const int col  = m0 + wc * 32 + ni * 16 + laneLo;
#pragma unroll
      for (int r = 0; r < 8; ++r)
        S[(size_t)(rowB + r) * M_DIM + col] = f32_to_bf16(acc[mi][ni][r]);
    }
}

// ---------------------------------------------------------------------------
// Row softmax in place over bf16 rows of length 8192 (one block per row)
// ---------------------------------------------------------------------------
__global__ __launch_bounds__(256) void softmax_inplace(unsigned short* __restrict__ S) {
  __shared__ float red[256];
  const int tid = threadIdx.x;
  unsigned short* rowp = S + (size_t)blockIdx.x * M_DIM;

  float x[32];
  const uint4* src = (const uint4*)(rowp + tid * 32);
#pragma unroll
  for (int j = 0; j < 4; ++j) {
    const uint4 q = src[j];
    const unsigned int w[4] = {q.x, q.y, q.z, q.w};
#pragma unroll
    for (int t = 0; t < 4; ++t) {
      x[j * 8 + t * 2 + 0] = bf16_to_f32((unsigned short)(w[t] & 0xFFFFu));
      x[j * 8 + t * 2 + 1] = bf16_to_f32((unsigned short)(w[t] >> 16));
    }
  }

  float m = -3.402823466e38f;
#pragma unroll
  for (int i = 0; i < 32; ++i) m = fmaxf(m, x[i]);
  red[tid] = m; __syncthreads();
  for (int s = 128; s > 0; s >>= 1) {
    if (tid < s) red[tid] = fmaxf(red[tid], red[tid + s]);
    __syncthreads();
  }
  m = red[0]; __syncthreads();

  float sum = 0.0f;
#pragma unroll
  for (int i = 0; i < 32; ++i) { x[i] = __expf(x[i] - m); sum += x[i]; }
  red[tid] = sum; __syncthreads();
  for (int s = 128; s > 0; s >>= 1) {
    if (tid < s) red[tid] += red[tid + s];
    __syncthreads();
  }
  const float inv = 1.0f / red[0];

  uint4* dst = (uint4*)(rowp + tid * 32);
#pragma unroll
  for (int j = 0; j < 4; ++j) {
    unsigned int w[4];
#pragma unroll
    for (int t = 0; t < 4; ++t) {
      const unsigned int lo = f32_to_bf16(x[j * 8 + t * 2 + 0] * inv);
      const unsigned int hi = f32_to_bf16(x[j * 8 + t * 2 + 1] * inv);
      w[t] = lo | (hi << 16);
    }
    dst[j] = make_uint4(w[0], w[1], w[2], w[3]);
  }
}

// ---------------------------------------------------------------------------
// GEMM2: O[d][v] = sum_m P[d][m] * bf16(V[v][m]),  O stored f32
// ---------------------------------------------------------------------------
__global__ __launch_bounds__(256) void gemm_pv(const unsigned short* __restrict__ P,
                                               const float* __restrict__ V,
                                               float* __restrict__ O) {
  __shared__ alignas(16) unsigned short As[2][128][40];   // [buf][d][mm]
  __shared__ alignas(16) unsigned short Bs[2][128][40];   // [buf][v][mm]

  const int tid    = threadIdx.x;
  const int lane   = tid & 31;
  const int laneLo = lane & 15;
  const int laneHi = lane >> 4;
  const int waveId = tid >> 5;
  const int wr     = waveId & 1;
  const int wc     = waveId >> 1;
  const int d0 = blockIdx.y * 128;
  const int n0 = blockIdx.x * 128;

  uint4  rp[2];
  float4 rv[4];
  // prologue: fetch K-slice 0
#pragma unroll
  for (int j = 0; j < 2; ++j) {
    const int idx = tid + 256 * j;
    const int row = idx >> 2;
    const int seg = idx & 3;
    rp[j] = *(const uint4*)(P + (size_t)(d0 + row) * M_DIM + seg * 8);
  }
#pragma unroll
  for (int j = 0; j < 4; ++j) {
    const int idx = tid + 256 * j;
    const int row = idx >> 3;
    const int seg = idx & 7;
    rv[j] = *(const float4*)(V + (size_t)(n0 + row) * M_DIM + seg * 4);
  }

  v8f acc[4][2] = {};
  int buf = 0;

  for (int k0 = 0; k0 < M_DIM; k0 += 32) {
    // commit prefetched registers to LDS
#pragma unroll
    for (int j = 0; j < 2; ++j) {
      const int idx = tid + 256 * j;
      const int row = idx >> 2;
      const int seg = idx & 3;
      *(uint4*)&As[buf][row][seg * 8] = rp[j];
    }
#pragma unroll
    for (int j = 0; j < 4; ++j) {
      const int idx = tid + 256 * j;
      const int row = idx >> 3;
      const int seg = idx & 7;
      const unsigned int lo = (unsigned int)f32_to_bf16(rv[j].x) |
                              ((unsigned int)f32_to_bf16(rv[j].y) << 16);
      const unsigned int hi = (unsigned int)f32_to_bf16(rv[j].z) |
                              ((unsigned int)f32_to_bf16(rv[j].w) << 16);
      *(uint2*)&Bs[buf][row][seg * 4] = make_uint2(lo, hi);
    }
    __syncthreads();

    // issue global loads for the next K-slice
    if (k0 + 32 < M_DIM) {
#pragma unroll
      for (int j = 0; j < 2; ++j) {
        const int idx = tid + 256 * j;
        const int row = idx >> 2;
        const int seg = idx & 3;
        rp[j] = *(const uint4*)(P + (size_t)(d0 + row) * M_DIM + k0 + 32 + seg * 8);
      }
#pragma unroll
      for (int j = 0; j < 4; ++j) {
        const int idx = tid + 256 * j;
        const int row = idx >> 3;
        const int seg = idx & 7;
        rv[j] = *(const float4*)(V + (size_t)(n0 + row) * M_DIM + k0 + 32 + seg * 4);
      }
    }

    Frag a[4], b[2];
#pragma unroll
    for (int mi = 0; mi < 4; ++mi) {
      const unsigned short* pa = &As[buf][wr * 64 + mi * 16 + laneLo][laneHi * 8];
      a[mi].q[0] = *(const uint4*)pa;
      a[mi].q[1] = *(const uint4*)(pa + 16);
    }
#pragma unroll
    for (int ni = 0; ni < 2; ++ni) {
      const unsigned short* pb = &Bs[buf][wc * 32 + ni * 16 + laneLo][laneHi * 8];
      b[ni].q[0] = *(const uint4*)pb;
      b[ni].q[1] = *(const uint4*)(pb + 16);
    }
#pragma unroll
    for (int mi = 0; mi < 4; ++mi)
#pragma unroll
      for (int ni = 0; ni < 2; ++ni)
        acc[mi][ni] = __builtin_amdgcn_wmma_f32_16x16x32_bf16(
            false, a[mi].v, false, b[ni].v, (short)0, acc[mi][ni], false, false);

    buf ^= 1;
  }

#pragma unroll
  for (int mi = 0; mi < 4; ++mi)
#pragma unroll
    for (int ni = 0; ni < 2; ++ni) {
      const int rowB = d0 + wr * 64 + mi * 16 + 8 * laneHi;
      const int col  = n0 + wc * 32 + ni * 16 + laneLo;
#pragma unroll
      for (int r = 0; r < 8; ++r)
        O[(size_t)(rowB + r) * V_DIM + col] = acc[mi][ni][r];
    }
}

// ---------------------------------------------------------------------------
extern "C" void kernel_launch(void* const* d_in, const int* in_sizes, int n_in,
                              void* d_out, int out_size, void* d_ws, size_t ws_size,
                              hipStream_t stream) {
  const float* Q = (const float*)d_in[0];   // (4096, 4096)
  const float* K = (const float*)d_in[1];   // (4096, 8192)
  const float* V = (const float*)d_in[2];   // (4096, 8192)
  float* O = (float*)d_out;                 // (4096, 4096)
  unsigned short* S = (unsigned short*)d_ws;  // 4096*8192 bf16 = 64 MiB

  gemm_qk<<<dim3(M_DIM / 128, D_DIM / 128), 256, 0, stream>>>(Q, K, S);
  softmax_inplace<<<dim3(D_DIM), 256, 0, stream>>>(S);
  gemm_pv<<<dim3(V_DIM / 128, D_DIM / 128), 256, 0, stream>>>(S, V, O);
}
